// PairwiseRepresentation_62440234549494
// MI455X (gfx1250) — matrix operand order repaired and checked by reading, not compiled, not run on hardware
//
#include <hip/hip_runtime.h>
#include <hip/hip_bf16.h>
#include <stdint.h>

#define NRES 512
#define DIM  256
#define HIDC 128
#define OUTC 128
#define LNEPS 1e-5f

typedef __attribute__((ext_vector_type(16))) __bf16 v16bf;
typedef __attribute__((ext_vector_type(8)))  float  v8f;
typedef __attribute__((ext_vector_type(4)))  unsigned int v4u;

union FragB {
    v16bf bf;
    v4u   q[2];
};

__device__ __forceinline__ unsigned short f2bf(float f) {
    unsigned int u = __float_as_uint(f);
    u += 0x7FFFu + ((u >> 16) & 1u);   // round-to-nearest-even
    return (unsigned short)(u >> 16);
}

// ---------------------------------------------------------------------------
// Weight f32 -> bf16 conversion (weights are tiny; one-shot, L2 resident after)
// ---------------------------------------------------------------------------
__global__ void f32_to_bf16_kernel(const float* __restrict__ src,
                                   unsigned short* __restrict__ dst, int n) {
    int i = blockIdx.x * blockDim.x + threadIdx.x;
    if (i < n) dst[i] = f2bf(src[i]);
}

// ---------------------------------------------------------------------------
// Stage 1: input layernorm + left/right projections (f32, 0.13 GFLOP total)
// one block per residue n, 256 threads (one per D / PROJ channel)
// ---------------------------------------------------------------------------
__global__ __launch_bounds__(256) void stage1_ln_proj(
    const float* __restrict__ x,    const float* __restrict__ mask,
    const float* __restrict__ g_in, const float* __restrict__ o_in,
    const float* __restrict__ Wl,   const float* __restrict__ bl,
    const float* __restrict__ Wr,   const float* __restrict__ br,
    float* __restrict__ left, float* __restrict__ right)
{
    const int n = blockIdx.x;
    const int t = threadIdx.x;
    __shared__ float xn[DIM];
    __shared__ float red1[8], red2[8];

    float v  = x[n * DIM + t];
    float s1 = v, s2 = v * v;
    #pragma unroll
    for (int off = 16; off >= 1; off >>= 1) {
        s1 += __shfl_xor(s1, off, 32);
        s2 += __shfl_xor(s2, off, 32);
    }
    const int lane = t & 31, wv = t >> 5;
    if (lane == 0) { red1[wv] = s1; red2[wv] = s2; }
    __syncthreads();
    float tot1 = 0.f, tot2 = 0.f;
    #pragma unroll
    for (int i = 0; i < 8; ++i) { tot1 += red1[i]; tot2 += red2[i]; }
    const float mu   = tot1 * (1.0f / DIM);
    const float var  = tot2 * (1.0f / DIM) - mu * mu;
    const float rstd = rsqrtf(var + LNEPS);
    xn[t] = g_in[t] * (v - mu) * rstd + o_in[t];
    __syncthreads();

    float accL = bl[t], accR = br[t];
    #pragma unroll 4
    for (int d = 0; d < DIM; ++d) {
        const float xv = xn[d];
        accL = fmaf(xv, Wl[d * DIM + t], accL);   // coalesced column read
        accR = fmaf(xv, Wr[d * DIM + t], accR);
    }
    const float m = mask[n];
    left [n * DIM + t] = m * accL;
    right[n * DIM + t] = m * accR;
}

// ---------------------------------------------------------------------------
// Stage 2: per (n, 16-k-tile): pair outer-product tile -> two K=256 GEMMs
// (W1 + W_skip), relu, K=128 GEMM (W2), fused output layernorm.
// 8 waves/block, wave w owns output channels [16w,16w+16). 160 WMMAs/block.
// ---------------------------------------------------------------------------
__global__ __launch_bounds__(256) void stage2_pair(
    const float* __restrict__ left, const float* __restrict__ right,
    const unsigned short* __restrict__ W1b,
    const unsigned short* __restrict__ Wsb,
    const unsigned short* __restrict__ W2b,
    const float* __restrict__ b1,  const float* __restrict__ b2,
    const float* __restrict__ bsk,
    const float* __restrict__ gam, const float* __restrict__ bet,
    float* __restrict__ out)
{
    // row strides padded so 16 rows spread across all 64 LDS banks
    const int PA_STR = 264;   // ushort stride for 16x256 bf16 pair tile
    const int HA_STR = 136;   // ushort stride for 16x128 bf16 hid tile
    const int OT_STR = 132;   // float  stride for 16x128 f32 out tile
    __shared__ __align__(16) unsigned short pairA[16 * PA_STR];
    __shared__ __align__(16) unsigned short hidA [16 * HA_STR];
    __shared__ __align__(16) float          outT [16 * OT_STR];

    const int n   = blockIdx.y;
    const int k0  = blockIdx.x * 16;
    const int tid = threadIdx.x;

    // ---- build bf16 pair tile: pairA[row][d] = left[n,d] * right[k0+row,d]
    for (int idx = tid; idx < 16 * DIM; idx += 256) {
        const int row = idx >> 8;          // / 256
        const int d   = idx & (DIM - 1);
        const float pv = left[n * DIM + d] * right[(k0 + row) * DIM + d];
        pairA[row * PA_STR + d] = f2bf(pv);
    }
    __syncthreads();

    const int wave = tid >> 5;             // output-channel tile id
    const int lane = tid & 31;
    const int hhi  = lane >> 4;            // lane half (A K-split / C row-split)
    const int mrow = lane & 15;            // A row this lane supplies
    const int coln = lane & 15;            // C/D column this lane holds
    const int oc   = wave * 16;            // output-channel base

    v8f acc1 = {};   // hid pre-activation accumulator
    v8f accS = {};   // skip accumulator (later also receives hid @ W2)

    // ---- GEMM 1+2: contract d=0..255 against W1 and W_skip (bf16 WMMA)
    #pragma unroll
    for (int c = 0; c < DIM / 32; ++c) {
        FragB a, bw, bs;
        const unsigned short* ap = &pairA[mrow * PA_STR + c * 32 + hhi * 8];
        a.q[0] = *reinterpret_cast<const v4u*>(ap);        // K = base..base+7
        a.q[1] = *reinterpret_cast<const v4u*>(ap + 16);   // K = base+16..+23
        #pragma unroll
        for (int s = 0; s < 16; ++s) {
            const int kk = c * 32 + hhi * 8 + (s < 8 ? s : s + 8);
            bw.bf[s] = __builtin_bit_cast(__bf16, W1b[kk * HIDC + oc + coln]);
            bs.bf[s] = __builtin_bit_cast(__bf16, Wsb[kk * OUTC + oc + coln]);
        }
        acc1 = __builtin_amdgcn_wmma_f32_16x16x32_bf16(
                   false, a.bf, false, bw.bf, (short)0, acc1, false, false);
        accS = __builtin_amdgcn_wmma_f32_16x16x32_bf16(
                   false, a.bf, false, bs.bf, (short)0, accS, false, false);
    }

    // ---- bias + relu, publish hid tile (bf16) for the K=128 contraction
    const float b1v = b1[oc + coln];
    #pragma unroll
    for (int r = 0; r < 8; ++r) {
        float h = acc1[r] + b1v;
        h = h > 0.f ? h : 0.f;
        hidA[(r + 8 * hhi) * HA_STR + oc + coln] = f2bf(h);
    }
    __syncthreads();

    // ---- GEMM 3: hid @ W2, accumulated straight into the skip accumulator
    #pragma unroll
    for (int c = 0; c < HIDC / 32; ++c) {
        FragB a, bw;
        const unsigned short* ap = &hidA[mrow * HA_STR + c * 32 + hhi * 8];
        a.q[0] = *reinterpret_cast<const v4u*>(ap);
        a.q[1] = *reinterpret_cast<const v4u*>(ap + 16);
        #pragma unroll
        for (int s = 0; s < 16; ++s) {
            const int kk = c * 32 + hhi * 8 + (s < 8 ? s : s + 8);
            bw.bf[s] = __builtin_bit_cast(__bf16, W2b[kk * OUTC + oc + coln]);
        }
        accS = __builtin_amdgcn_wmma_f32_16x16x32_bf16(
                   false, a.bf, false, bw.bf, (short)0, accS, false, false);
    }

    const float b2v = b2[oc + coln] + bsk[oc + coln];
    #pragma unroll
    for (int r = 0; r < 8; ++r)
        outT[(r + 8 * hhi) * OT_STR + oc + coln] = accS[r] + b2v;
    __syncthreads();

    // ---- fused output layernorm over the 128 channels of each pair row
    const int row = tid >> 4;              // 16 rows, 16 threads each
    const int g   = tid & 15;
    float s1 = 0.f, s2 = 0.f;
    #pragma unroll
    for (int j = 0; j < 8; ++j) {
        const float v = outT[row * OT_STR + g + 16 * j];
        s1 += v; s2 += v * v;
    }
    #pragma unroll
    for (int off = 8; off >= 1; off >>= 1) {   // stays within 16-lane group
        s1 += __shfl_xor(s1, off, 32);
        s2 += __shfl_xor(s2, off, 32);
    }
    const float mu   = s1 * (1.0f / OUTC);
    const float var  = s2 * (1.0f / OUTC) - mu * mu;
    const float rstd = rsqrtf(var + LNEPS);
    float* orow = out + ((size_t)n * NRES + (k0 + row)) * OUTC;
    #pragma unroll
    for (int j = 0; j < 8; ++j) {
        const int col = g + 16 * j;
        const float v = outT[row * OT_STR + col];
        orow[col] = gam[col] * (v - mu) * rstd + bet[col];
    }
}

// ---------------------------------------------------------------------------
extern "C" void kernel_launch(void* const* d_in, const int* in_sizes, int n_in,
                              void* d_out, int out_size, void* d_ws, size_t ws_size,
                              hipStream_t stream) {
    const float* x     = (const float*)d_in[0];
    const float* mask  = (const float*)d_in[1];
    const float* lni_s = (const float*)d_in[2];
    const float* lni_o = (const float*)d_in[3];
    const float* Wl    = (const float*)d_in[4];
    const float* bl    = (const float*)d_in[5];
    const float* Wr    = (const float*)d_in[6];
    const float* br    = (const float*)d_in[7];
    const float* W1    = (const float*)d_in[8];
    const float* b1    = (const float*)d_in[9];
    const float* W2    = (const float*)d_in[10];
    const float* b2    = (const float*)d_in[11];
    const float* Wsk   = (const float*)d_in[12];
    const float* bsk   = (const float*)d_in[13];
    const float* lno_s = (const float*)d_in[14];
    const float* lno_o = (const float*)d_in[15];
    float* out = (float*)d_out;

    // workspace layout (~1.16 MB total)
    char* ws = (char*)d_ws;
    float* left  = (float*)(ws);
    float* right = (float*)(ws + (size_t)NRES * DIM * 4);
    unsigned short* W1b = (unsigned short*)(ws + (size_t)2 * NRES * DIM * 4);
    unsigned short* Wsb = W1b + DIM * HIDC;
    unsigned short* W2b = Wsb + DIM * OUTC;

    f32_to_bf16_kernel<<<(DIM * HIDC + 255) / 256, 256, 0, stream>>>(W1,  W1b, DIM * HIDC);
    f32_to_bf16_kernel<<<(DIM * OUTC + 255) / 256, 256, 0, stream>>>(Wsk, Wsb, DIM * OUTC);
    f32_to_bf16_kernel<<<(HIDC * OUTC + 255) / 256, 256, 0, stream>>>(W2,  W2b, HIDC * OUTC);

    stage1_ln_proj<<<NRES, 256, 0, stream>>>(x, mask, lni_s, lni_o,
                                             Wl, bl, Wr, br, left, right);

    dim3 grid(NRES / 16, NRES);   // (k-tiles, n)
    stage2_pair<<<grid, 256, 0, stream>>>(left, right, W1b, Wsb, W2b,
                                          b1, b2, bsk, lno_s, lno_o, out);
}
